// LocalSphereAttention_56040733278768
// MI455X (gfx1250) — matrix operand order repaired and checked by reading, not compiled, not run on hardware
//
#include <hip/hip_runtime.h>
#include <stdint.h>

typedef __attribute__((ext_vector_type(16))) __bf16 v16bf;
typedef __attribute__((ext_vector_type(8)))  __bf16 v8bf;
typedef __attribute__((ext_vector_type(8)))  float  v8f;
typedef __attribute__((ext_vector_type(4)))  unsigned int su4;
typedef __attribute__((ext_vector_type(8)))  unsigned int su8;

#define BB       2
#define NPTS     8192
#define DIMC     256
#define NHEAD    8
#define KNBR     32
#define HDIM     32
#define M_TOTAL  (BB * NPTS)   // 16384 rows

static __device__ __forceinline__ unsigned short f32_to_bf16(float f) {
  unsigned u = __float_as_uint(f);
  u += 0x7FFFu + ((u >> 16) & 1u);           // round-to-nearest-even
  return (unsigned short)(u >> 16);
}

// ---------------- conversion / setup kernels ----------------

__global__ void cvt_f32_bf16(const float* __restrict__ in,
                             unsigned short* __restrict__ out, int n) {
  int i = blockIdx.x * blockDim.x + threadIdx.x;
  if (i < n) out[i] = f32_to_bf16(in[i]);
}

// Transpose+convert the four 256x256 weight matrices: W[k][n] -> Wt[n][k] (bf16).
__global__ void cvt_weights(const float* __restrict__ Wq, const float* __restrict__ Wk,
                            const float* __restrict__ Wv, const float* __restrict__ Wo,
                            unsigned short* __restrict__ WqkvT,
                            unsigned short* __restrict__ WoT) {
  int k = blockIdx.x;    // 0..255  row of W
  int n = threadIdx.x;   // 0..255  col of W
  WqkvT[(size_t)(0 * DIMC + n) * DIMC + k] = f32_to_bf16(Wq[k * DIMC + n]);
  WqkvT[(size_t)(1 * DIMC + n) * DIMC + k] = f32_to_bf16(Wk[k * DIMC + n]);
  WqkvT[(size_t)(2 * DIMC + n) * DIMC + k] = f32_to_bf16(Wv[k * DIMC + n]);
  WoT  [(size_t)n * DIMC + k]              = f32_to_bf16(Wo[k * DIMC + n]);
}

__global__ void cat_bias(const float* __restrict__ bq, const float* __restrict__ bk,
                         const float* __restrict__ bv, float* __restrict__ out) {
  int i = blockIdx.x * blockDim.x + threadIdx.x;   // 0..767
  float v = (i < 256) ? bq[i] : ((i < 512) ? bk[i - 256] : bv[i - 512]);
  out[i] = v;
}

// ---------------- A-tile staging: TDM descriptor path or async-copy path ----------------
// Stages a 64(M) x 32(K) bf16 tile (4 KB) of A into LDS at ldsdst.

template <int USE_TDM>
static __device__ __forceinline__ void stage_tile(
    unsigned long long abase,          // byte address of A
    unsigned short* ldsdst,            // LDS destination (tile base)
    int mbase, int k0, int Kdim,
    int tid, int wid) {
  if (USE_TDM) {
    // One Tensor Data Mover descriptor, issued by wave 0 only.
    // tensor_load_to_lds ignores EXEC, so the gate must be a *scalar* branch:
    // readfirstlane(wid) forces an SGPR compare -> s_cbranch, never exec-masking.
    if (__builtin_amdgcn_readfirstlane(wid) == 0) {
      unsigned ldsb = (unsigned)(uintptr_t)ldsdst;
      unsigned long long ga = abase + ((unsigned long long)((size_t)mbase * Kdim + k0) << 1);
      su4 g0;
      g0[0] = 1u;                                           // count=1, is_restore=0
      g0[1] = ldsb;                                         // lds_addr
      g0[2] = (unsigned)ga;                                 // global_addr[31:0]
      g0[3] = ((unsigned)(ga >> 32) & 0x01FFFFFFu) | (2u << 30);  // addr[56:32] | type=2
      su8 g1;
      g1[0] = (1u << 16);                                   // workgroup_mask=0, data_size=2B
      g1[1] = ((unsigned)Kdim & 0xFFFFu) << 16;             // tensor_dim0[15:0]
      g1[2] = ((unsigned)Kdim >> 16) | (((unsigned)M_TOTAL & 0xFFFFu) << 16); // td0 hi | td1 lo
      g1[3] = ((unsigned)M_TOTAL >> 16) | (32u << 16);      // td1 hi | tile_dim0=32
      g1[4] = 64u;                                          // tile_dim1=64, tile_dim2=0
      g1[5] = (unsigned)Kdim;                               // tensor_dim0_stride[31:0]
      g1[6] = 0u;                                           // stride0 hi | stride1 lo
      g1[7] = 0u;
      asm volatile("tensor_load_to_lds %0, %1" :: "s"(g0), "s"(g1) : "memory");
    }
  } else {
    // Per-thread async copy: 256 threads x b128 = 4 KB.
    int row = tid >> 2;                        // 0..63
    int ch  = tid & 3;                         // 0..3 (8 bf16 each)
    unsigned lds  = (unsigned)(uintptr_t)(ldsdst + row * 32 + ch * 8);
    unsigned goff = (unsigned)((((size_t)(mbase + row)) * Kdim + (k0 + ch * 8)) * 2);
    asm volatile("global_load_async_to_lds_b128 %0, %1, %2"
                 :: "v"(lds), "v"(goff), "s"(abase) : "memory");
  }
}

template <int USE_TDM>
static __device__ __forceinline__ void wait_stage(int wid) {
  if (USE_TDM) {
    if (__builtin_amdgcn_readfirstlane(wid) == 0)
      __builtin_amdgcn_s_wait_tensorcnt(0);
  } else {
    asm volatile("s_wait_asynccnt 0" ::: "memory");
  }
}

// ---------------- WMMA GEMM:  C[M,N] = A[M,K] * Bt[N,K]^T + bias ----------------
// A: bf16 row-major [M,Kdim].  Bt: bf16 [N,Kdim] (B transposed -> per-lane K-contiguous).
// Workgroup: 256 threads = 8 waves; tile 64(M) x 256(N); wave = 32x64 (2x4 WMMA tiles).
// A tiles double-buffered in LDS; staging (TDM or async) overlaps the 8 WMMAs per step.

template <int USE_TDM>
__global__ __launch_bounds__(256) void gemm_bf16_wmma(
    const unsigned short* __restrict__ A,
    const unsigned short* __restrict__ Bt,
    const float* __restrict__ bias,
    float* __restrict__ C,
    int Kdim, int Nld) {
  __shared__ unsigned short tileA[2][64 * 32];   // 2 x 4 KB, double buffered

  const int mbase = blockIdx.x * 64;
  const int nbase = blockIdx.y * 256;
  const int tid   = threadIdx.x;
  const int wid   = tid >> 5;
  const int lane  = tid & 31;
  const int wm    = wid >> 2;                    // 0..1 -> +32*wm rows
  const int wn    = wid & 3;                     // 0..3 -> +64*wn cols
  const int lrow  = lane & 15;
  const int lhi   = lane >> 4;                   // 0/1 half-wave

  v8f acc[2][4];
#pragma unroll
  for (int i = 0; i < 2; ++i)
#pragma unroll
    for (int j = 0; j < 4; ++j)
#pragma unroll
      for (int r = 0; r < 8; ++r) acc[i][j][r] = 0.0f;

  const unsigned long long abase = (unsigned long long)(uintptr_t)A;
  const int nsteps = Kdim >> 5;                  // K-steps of 32

  // prologue: stage first tile into buffer 0
  stage_tile<USE_TDM>(abase, &tileA[0][0], mbase, 0, Kdim, tid, wid);

  for (int s = 0; s < nsteps; ++s) {
    const int p  = s & 1;
    const int k0 = s << 5;

    wait_stage<USE_TDM>(wid);                    // copy into buf p complete (this wave)
    __syncthreads();                             // visible to whole workgroup

    if (s + 1 < nsteps)                          // overlap next copy with compute
      stage_tile<USE_TDM>(abase, &tileA[1 - p][0], mbase, k0 + 32, Kdim, tid, wid);

    // prefetch a future B stripe into L2
    __builtin_prefetch(Bt + ((size_t)(nbase + wn * 64 + lrow) * Kdim + ((k0 + 32) & (Kdim - 1))), 0, 1);

    // ---- A fragments (contiguous K = lhi*16+{0..15}, matching B) ----
    v16bf af[2];
#pragma unroll
    for (int i = 0; i < 2; ++i) {
      int r = wm * 32 + i * 16 + lrow;
      union { v16bf v; v8bf h[2]; } u;
      u.v = *(const v16bf*)&tileA[p][r * 32 + lhi * 16];
      af[i] = u.v;
    }

    // ---- B fragments from global Wt (L2-resident) + 8 WMMAs ----
#pragma unroll
    for (int j = 0; j < 4; ++j) {
      int n = nbase + wn * 64 + j * 16 + lrow;
      v16bf bfrag = *(const v16bf*)&Bt[(size_t)n * Kdim + k0 + lhi * 16];
#pragma unroll
      for (int i = 0; i < 2; ++i)
        acc[i][j] = __builtin_amdgcn_wmma_f32_16x16x32_bf16(
            false, af[i], false, bfrag, (short)0, acc[i][j], false, false);
    }
    __syncthreads();                             // all reads of buf p done before overwrite
  }

  // ---- epilogue: C/D layout -> lane holds col lane%16, VGPR r holds row r+8*lhi ----
#pragma unroll
  for (int i = 0; i < 2; ++i)
#pragma unroll
    for (int j = 0; j < 4; ++j) {
      int n = nbase + wn * 64 + j * 16 + lrow;
      float bn = bias[n];
#pragma unroll
      for (int r = 0; r < 8; ++r) {
        int m = mbase + wm * 32 + i * 16 + r + 8 * lhi;
        C[(size_t)m * Nld + n] = acc[i][j][r] + bn;
      }
    }
}

// ---------------- attention core: 1 block per point, 1 wave per head ----------------
// lane = neighbor for score/softmax; lane = channel for V-accumulation (via LDS swap).

__global__ __launch_bounds__(256) void attn_kernel(
    const float* __restrict__ qkv,        // [M, 768]  q|k|v
    const float* __restrict__ xyz,        // [B, N, 3]
    const int*   __restrict__ idx,        // [B, N, 32]
    const float* __restrict__ W1, const float* __restrict__ b1,   // [3,32],[32]
    const float* __restrict__ W2, const float* __restrict__ b2,   // [32,8],[8]
    unsigned short* __restrict__ attn_out /* [M, 256] bf16 */) {
  __shared__ float s_attn[NHEAD][KNBR];
  __shared__ int   s_nb[NHEAD][KNBR];

  const int p    = blockIdx.x;            // point row 0..16383
  const int b    = p >> 13;               // / 8192
  const int h    = threadIdx.x >> 5;      // head
  const int lane = threadIdx.x & 31;      // neighbor index j

  const int nb  = idx[(size_t)p * KNBR + lane];
  const int nbg = b * NPTS + nb;          // neighbor's global row

  // relative position
  float rx = xyz[(size_t)p * 3 + 0] - xyz[(size_t)nbg * 3 + 0];
  float ry = xyz[(size_t)p * 3 + 1] - xyz[(size_t)nbg * 3 + 1];
  float rz = xyz[(size_t)p * 3 + 2] - xyz[(size_t)nbg * 3 + 2];

  // bias MLP: relu(rel @ W1 + b1) @ W2 + b2, only this wave's head column
  float biasv = b2[h];
#pragma unroll
  for (int i = 0; i < 32; ++i) {
    float hsum = fmaf(rx, W1[0 * 32 + i],
                 fmaf(ry, W1[1 * 32 + i],
                 fmaf(rz, W1[2 * 32 + i], b1[i])));
    hsum = fmaxf(hsum, 0.0f);
    biasv = fmaf(hsum, W2[i * NHEAD + h], biasv);
  }

  // q . k_neighbor over HD=32, q broadcast through wave32 shuffles
  const float* qrow = qkv + (size_t)p * 768 + h * HDIM;
  const float* krow = qkv + (size_t)nbg * 768 + 256 + h * HDIM;
  float qreg  = qrow[lane];
  float score = 0.0f;
#pragma unroll
  for (int d = 0; d < HDIM; ++d) {
    float qd = __shfl(qreg, d, 32);
    score = fmaf(qd, krow[d], score);
  }
  score = fmaf(score, 0.17677669529663687f, biasv);   // 1/sqrt(32)

  // softmax across the 32 lanes (neighbors)
  float mx = score;
#pragma unroll
  for (int off = 16; off > 0; off >>= 1) mx = fmaxf(mx, __shfl_xor(mx, off, 32));
  float e = __expf(score - mx);
  float s = e;
#pragma unroll
  for (int off = 16; off > 0; off >>= 1) s += __shfl_xor(s, off, 32);
  float attn = e / s;

  s_attn[h][lane] = attn;
  s_nb[h][lane]   = nbg;
  __syncthreads();

  // lane = channel d: out[d] = sum_j attn_j * v[nb_j][h*32+d]  (coalesced across lanes)
  float accv = 0.0f;
#pragma unroll
  for (int j = 0; j < KNBR; ++j) {
    const float* vrow = qkv + (size_t)s_nb[h][j] * 768 + 512 + h * HDIM;
    accv = fmaf(s_attn[h][j], vrow[lane], accv);
  }
  attn_out[(size_t)p * DIMC + h * HDIM + lane] = f32_to_bf16(accv);
}

// ---------------- launch ----------------

extern "C" void kernel_launch(void* const* d_in, const int* in_sizes, int n_in,
                              void* d_out, int out_size, void* d_ws, size_t ws_size,
                              hipStream_t stream) {
  const float* x   = (const float*)d_in[0];
  const float* xyz = (const float*)d_in[1];
  const int*   idx = (const int*)d_in[2];
  const float* Wq  = (const float*)d_in[3];
  const float* bq  = (const float*)d_in[4];
  const float* Wk  = (const float*)d_in[5];
  const float* bk  = (const float*)d_in[6];
  const float* Wv  = (const float*)d_in[7];
  const float* bv  = (const float*)d_in[8];
  const float* Wo  = (const float*)d_in[9];
  const float* bo  = (const float*)d_in[10];
  const float* W1  = (const float*)d_in[11];
  const float* b1  = (const float*)d_in[12];
  const float* W2  = (const float*)d_in[13];
  const float* b2  = (const float*)d_in[14];
  float* out = (float*)d_out;

  char* ws = (char*)d_ws;
  size_t off = 0;
  auto alloc = [&](size_t bytes) -> char* {
    char* p = ws + off;
    off += (bytes + 255) & ~(size_t)255;
    return p;
  };
  unsigned short* x_bf    = (unsigned short*)alloc((size_t)M_TOTAL * DIMC * 2);   // 8.4 MB
  unsigned short* WqkvT   = (unsigned short*)alloc((size_t)768 * DIMC * 2);       // 384 KB
  unsigned short* WoT     = (unsigned short*)alloc((size_t)DIMC * DIMC * 2);      // 128 KB
  float*          biascat = (float*)alloc(768 * sizeof(float));
  float*          qkv     = (float*)alloc((size_t)M_TOTAL * 768 * sizeof(float)); // 50 MB
  unsigned short* attn_bf = (unsigned short*)alloc((size_t)M_TOTAL * DIMC * 2);   // 8.4 MB
  (void)ws_size; (void)in_sizes; (void)n_in; (void)out_size;

  // 1) fp32 -> bf16 activations; transpose+convert weights; concat q/k/v bias
  cvt_f32_bf16<<<(M_TOTAL * DIMC + 255) / 256, 256, 0, stream>>>(x, x_bf, M_TOTAL * DIMC);
  cvt_weights<<<DIMC, DIMC, 0, stream>>>(Wq, Wk, Wv, Wo, WqkvT, WoT);
  cat_bias<<<3, 256, 0, stream>>>(bq, bk, bv, biascat);

  // 2) fused QKV projection: [16384,256] x [256,768] via WMMA bf16, A tiles via TDM
  gemm_bf16_wmma<1><<<dim3(M_TOTAL / 64, 3), 256, 0, stream>>>(x_bf, WqkvT, biascat, qkv, DIMC, 768);

  // 3) neighborhood attention (gathers are L2-resident: qkv is ~50 MB < 192 MB L2)
  attn_kernel<<<M_TOTAL, 256, 0, stream>>>(qkv, xyz, idx, W1, b1, W2, b2, attn_bf);

  // 4) output projection via WMMA bf16, A tiles via async-to-LDS copies
  gemm_bf16_wmma<0><<<dim3(M_TOTAL / 64, 1), 256, 0, stream>>>(attn_bf, WoT, bo, out, DIMC, DIMC);
}
